// Transformer_32203664785692
// MI455X (gfx1250) — matrix operand order repaired and checked by reading, not compiled, not run on hardware
//
#include <hip/hip_runtime.h>
#include <hip/hip_bf16.h>

typedef __attribute__((ext_vector_type(16))) _Float16 v16h;
typedef __attribute__((ext_vector_type(8)))  float    v8f;
typedef _Float16 half_t;

#define NEG_BIG (-1000000000.0f)

static inline int ceilDiv(long long a, long long b) { return (int)((a + b - 1) / b); }

// ---------------------------------------------------------------------------
// Embedding + sinusoidal positional encoding
// ---------------------------------------------------------------------------
__global__ void embed_pe_kernel(const float* __restrict__ emb, const int* __restrict__ ids,
                                float* __restrict__ out, int Bn, int Ltok, int Luse, int Dm) {
  long long idx = (long long)blockIdx.x * blockDim.x + threadIdx.x;
  long long total = (long long)Bn * Luse * Dm;
  if (idx >= total) return;
  int d = (int)(idx % Dm);
  long long t2 = idx / Dm;
  int t = (int)(t2 % Luse);
  int b = (int)(t2 / Luse);
  int id = ids[(long long)b * Ltok + t];
  int pair = d >> 1;
  float expo = (float)(2 * pair) / (float)Dm;
  float ang = (float)t * powf(10000.0f, -expo);
  float pe = (d & 1) ? cosf(ang) : sinf(ang);
  out[idx] = emb[(long long)id * Dm + d] + pe;
}

// ---------------------------------------------------------------------------
// f32 [rows x cols] row-major -> f16 [rows x Kp] row-major, zero-pad cols..Kp
// ---------------------------------------------------------------------------
__global__ void cvt_a_kernel(const float* __restrict__ src, half_t* __restrict__ dst,
                             long long rows, int cols, int Kp) {
  long long idx = (long long)blockIdx.x * blockDim.x + threadIdx.x;
  long long total = rows * (long long)Kp;
  if (idx >= total) return;
  int c = (int)(idx % Kp);
  long long r = idx / Kp;
  float v = (c < cols) ? src[r * cols + c] : 0.0f;
  dst[idx] = (half_t)v;
}

// ---------------------------------------------------------------------------
// Weight f32 [K x N] -> transposed padded f16 Bt [Np x Kp]  (Np mult of 64)
// ---------------------------------------------------------------------------
__global__ void cvt_w_t_kernel(const float* __restrict__ w, half_t* __restrict__ dst,
                               int K, int N, int Kp, int Np) {
  long long idx = (long long)blockIdx.x * blockDim.x + threadIdx.x;
  long long total = (long long)Np * Kp;
  if (idx >= total) return;
  int k = (int)(idx % Kp);
  int n = (int)(idx / Kp);
  float v = (k < K && n < N) ? w[(long long)k * N + n] : 0.0f;
  dst[idx] = (half_t)v;
}

// ---------------------------------------------------------------------------
// Per-head staging (row-major A / K side):
// dst[((b*H + h)*Lp + r)*HDp + c] = src[(b*L + r)*srcld + h*HD + c]
// ---------------------------------------------------------------------------
__global__ void stage_heads_kernel(const float* __restrict__ src, half_t* __restrict__ dst,
                                   int Bn, int Hn, int L, int Lp, int srcld, int HD, int HDp) {
  long long idx = (long long)blockIdx.x * blockDim.x + threadIdx.x;
  long long total = (long long)Bn * Hn * Lp * HDp;
  if (idx >= total) return;
  int c = (int)(idx % HDp);
  long long t = idx / HDp;
  int r = (int)(t % Lp); t /= Lp;
  int h = (int)(t % Hn);
  int b = (int)(t / Hn);
  float v = 0.0f;
  if (c < HD && r < L) v = src[((long long)b * L + r) * srcld + h * HD + c];
  dst[idx] = (half_t)v;
}

// ---------------------------------------------------------------------------
// Per-head V transposed staging (Bt side for A@V):
// dst[((b*H + h)*HDp + n)*Lkp + k] = src[(b*L + k)*srcld + h*HD + n]
// ---------------------------------------------------------------------------
__global__ void stage_heads_t_kernel(const float* __restrict__ src, half_t* __restrict__ dst,
                                     int Bn, int Hn, int L, int Lkp, int srcld, int HD, int HDp) {
  long long idx = (long long)blockIdx.x * blockDim.x + threadIdx.x;
  long long total = (long long)Bn * Hn * HDp * Lkp;
  if (idx >= total) return;
  int k = (int)(idx % Lkp);
  long long t = idx / Lkp;
  int n = (int)(t % HDp); t /= HDp;
  int h = (int)(t % Hn);
  int b = (int)(t / Hn);
  float v = 0.0f;
  if (n < HD && k < L) v = src[((long long)b * L + k) * srcld + h * HD + n];
  dst[idx] = (half_t)v;
}

// ---------------------------------------------------------------------------
// scores f32 [Z x 256 x 256] -> f16 same layout, zero where q>=Lq or k>=Lk
// ---------------------------------------------------------------------------
__global__ void cvt_scores16_kernel(const float* __restrict__ src, half_t* __restrict__ dst,
                                    long long total, int Lq, int Lk) {
  long long idx = (long long)blockIdx.x * blockDim.x + threadIdx.x;
  if (idx >= total) return;
  int k = (int)(idx % 256);
  int q = (int)((idx / 256) % 256);
  float v = (q < Lq && k < Lk) ? src[idx] : 0.0f;
  dst[idx] = (half_t)v;
}

// ---------------------------------------------------------------------------
// Batched WMMA GEMM: one wave computes a 16x64 tile (4 accumulators).
// A row-major padded to Kp (mult of 32). Bt stores B transposed with rows
// padded to mult of 64: elem(k,n) = Bt[n*ldbt + k] (contiguous in k).
// Per K-step: 1 shared A fragment + 4 B fragments -> 4 back-to-back WMMAs.
// ---------------------------------------------------------------------------
__global__ void wmma_gemm_kernel(const half_t* __restrict__ A, long long strideA, int lda,
                                 const half_t* __restrict__ Bt, long long strideB, int ldbt,
                                 float* __restrict__ C, long long strideC, int ldc,
                                 int M, int N, int Kp,
                                 const float* __restrict__ bias,
                                 float scale, int gelu, int causal,
                                 int headH, int headColOff) {
  int wid = threadIdx.x >> 5;
  int lane = threadIdx.x & 31;
  int tilesM = (M + 15) >> 4;
  int tilesN64 = (N + 63) >> 6;
  int tid = blockIdx.x * (blockDim.x >> 5) + wid;
  if (tid >= tilesM * tilesN64) return;
  int mt = tid % tilesM;
  int nt4 = tid / tilesM;
  int z = blockIdx.z;

  const half_t* Az = A + (long long)z * strideA;
  const half_t* Bz = Bt + (long long)z * strideB;
  float* Cz;
  if (headH > 0)
    Cz = C + (long long)(z / headH) * strideC + (long long)(z % headH) * headColOff;
  else
    Cz = C + (long long)z * strideC;

  int laneL = lane & 15;
  int hi = lane >> 4;  // 0 for lanes 0-15, 1 for lanes 16-31
  int am = mt * 16 + laneL;
  int bnBase = nt4 * 64 + laneL;

  v8f acc[4];
#pragma unroll
  for (int j = 0; j < 4; ++j) acc[j] = {};

  for (int k0 = 0; k0 < Kp; k0 += 32) {
    // A fragment (16-bit A 16x32 layout):
    //  lanes 0-15 : row m, K = k0+{0..7} then k0+{16..23}
    //  lanes 16-31: row m, K = k0+{8..15} then k0+{24..31}
    const half_t* ap = Az + (long long)am * lda + k0 + hi * 8;
    v16h a;
#pragma unroll
    for (int i = 0; i < 8; ++i) { a[i] = ap[i]; a[i + 8] = ap[i + 16]; }
    __builtin_prefetch(ap + 32, 0, 3);

    // 4 B fragments (16-bit B 32x16 layout), columns bnBase + 16*j
    v16h b[4];
#pragma unroll
    for (int j = 0; j < 4; ++j) {
      const half_t* bp = Bz + (long long)(bnBase + 16 * j) * ldbt + k0 + hi * 8;
#pragma unroll
      for (int i = 0; i < 8; ++i) { b[j][i] = bp[i]; b[j][i + 8] = bp[i + 16]; }
      __builtin_prefetch(bp + 32, 0, 3);
    }
#pragma unroll
    for (int j = 0; j < 4; ++j) {
      acc[j] = __builtin_amdgcn_wmma_f32_16x16x32_f16(
          /*neg_a=*/false, a, /*neg_b=*/false, b[j],
          /*c_mod=*/(short)0, acc[j], /*reuse_a=*/false, /*reuse_b=*/false);
    }
  }

  int rbase = mt * 16 + hi * 8;
#pragma unroll
  for (int j = 0; j < 4; ++j) {
    int col = nt4 * 64 + 16 * j + laneL;
    if (col < N) {
      float bb = bias ? bias[col] : 0.0f;
#pragma unroll
      for (int r = 0; r < 8; ++r) {
        int row = rbase + r;
        if (row < M) {
          float v = acc[j][r] * scale + bb;
          if (causal && col > row) v += NEG_BIG;
          if (gelu) v = 0.5f * v * (1.0f + erff(v * 0.70710678118654752f));
          Cz[(long long)row * ldc + col] = v;
        }
      }
    }
  }
}

// ---------------------------------------------------------------------------
// Row softmax over scores buffer [Z x 256 x 256]
// ---------------------------------------------------------------------------
__global__ void softmax_rows_kernel(float* __restrict__ sc, int Lq, int Lk) {
  int q = blockIdx.x;
  int z = blockIdx.y;
  if (q >= Lq) return;
  float* row = sc + ((long long)z * 256 + q) * 256;
  int t = threadIdx.x;
  __shared__ float red[256];
  float x = (t < Lk) ? row[t] : -3.0e38f;
  red[t] = x;
  __syncthreads();
  for (int s = 128; s > 0; s >>= 1) {
    if (t < s) red[t] = fmaxf(red[t], red[t + s]);
    __syncthreads();
  }
  float m = red[0];
  __syncthreads();
  float e = (t < Lk) ? expf(x - m) : 0.0f;
  red[t] = e;
  __syncthreads();
  for (int s = 128; s > 0; s >>= 1) {
    if (t < s) red[t] += red[t + s];
    __syncthreads();
  }
  float inv = 1.0f / red[0];
  if (t < Lk) row[t] = e * inv;
}

// ---------------------------------------------------------------------------
// LayerNorm per row of length Dm
// ---------------------------------------------------------------------------
__global__ void layernorm_kernel(const float* __restrict__ x, const float* __restrict__ g,
                                 const float* __restrict__ bta, float* __restrict__ y, int Dm) {
  long long row = blockIdx.x;
  const float* px = x + row * Dm;
  float* py = y + row * Dm;
  int t = threadIdx.x;
  __shared__ float red[128];
  float s = 0.0f;
  for (int i = t; i < Dm; i += 128) s += px[i];
  red[t] = s;
  __syncthreads();
  for (int st = 64; st > 0; st >>= 1) {
    if (t < st) red[t] += red[t + st];
    __syncthreads();
  }
  float mean = red[0] / (float)Dm;
  __syncthreads();
  float v = 0.0f;
  for (int i = t; i < Dm; i += 128) {
    float d = px[i] - mean;
    v += d * d;
  }
  red[t] = v;
  __syncthreads();
  for (int st = 64; st > 0; st >>= 1) {
    if (t < st) red[t] += red[t + st];
    __syncthreads();
  }
  float rstd = rsqrtf(red[0] / (float)Dm + 1e-5f);
  for (int i = t; i < Dm; i += 128) py[i] = (px[i] - mean) * rstd * g[i] + bta[i];
}

__global__ void add_vec_kernel(const float* __restrict__ a, const float* __restrict__ b,
                               float* __restrict__ c, long long n) {
  long long idx = (long long)blockIdx.x * blockDim.x + threadIdx.x;
  if (idx < n) c[idx] = a[idx] + b[idx];
}

// ---------------------------------------------------------------------------
// In-place log-softmax per row of length N
// ---------------------------------------------------------------------------
__global__ void logsoftmax_kernel(float* __restrict__ x, int N) {
  long long row = blockIdx.x;
  float* p = x + row * (long long)N;
  int t = threadIdx.x;
  __shared__ float red[256];
  float m = -3.0e38f;
  for (int i = t; i < N; i += 256) m = fmaxf(m, p[i]);
  red[t] = m;
  __syncthreads();
  for (int s = 128; s > 0; s >>= 1) {
    if (t < s) red[t] = fmaxf(red[t], red[t + s]);
    __syncthreads();
  }
  m = red[0];
  __syncthreads();
  float sum = 0.0f;
  for (int i = t; i < N; i += 256) sum += expf(p[i] - m);
  red[t] = sum;
  __syncthreads();
  for (int s = 128; s > 0; s >>= 1) {
    if (t < s) red[t] += red[t + s];
    __syncthreads();
  }
  float lse = m + logf(red[0]);
  for (int i = t; i < N; i += 256) p[i] = p[i] - lse;
}

__global__ void zero_acc_kernel(float* acc) {
  if (blockIdx.x == 0 && threadIdx.x < 2) acc[threadIdx.x] = 0.0f;
}

__global__ void nll_kernel(const float* __restrict__ lp, const int* __restrict__ zh,
                           float* __restrict__ acc, int Bn, int Lz, int Ltok, int V) {
  long long idx = (long long)blockIdx.x * blockDim.x + threadIdx.x;
  long long total = (long long)Bn * Lz;
  if (idx >= total) return;
  int t = (int)(idx % Lz);
  int b = (int)(idx / Lz);
  int tgt = zh[(long long)b * Ltok + t + 1];
  float valid = (tgt != 0) ? 1.0f : 0.0f;
  float nll = -lp[idx * (long long)V + tgt];
  atomicAdd(&acc[0], nll * valid);
  atomicAdd(&acc[1], valid);
}

__global__ void finish_loss_kernel(const float* __restrict__ acc, float* __restrict__ out) {
  if (blockIdx.x == 0 && threadIdx.x == 0) out[0] = acc[0] / fmaxf(acc[1], 1.0f);
}

// ---------------------------------------------------------------------------
// Host orchestration
// ---------------------------------------------------------------------------
extern "C" void kernel_launch(void* const* d_in, const int* in_sizes, int n_in,
                              void* d_out, int out_size, void* d_ws, size_t ws_size,
                              hipStream_t stream) {
  (void)in_sizes; (void)n_in; (void)out_size; (void)ws_size;

  const int Bsz = 16, Hn = 3, Dm = 300, Dp = 320, HD = 100, HDp = 128;
  const int Le = 256, Lz = 255;
  const int ME = Bsz * Le;   // 4096
  const int MZ = Bsz * Lz;   // 4080
  const int DFE = 1024, DFD = 512;
  const int V = 12016;
  const int Vp64 = 12032;    // vocab rows padded to x64 for the Bt arena
  const int ZBH = Bsz * Hn;  // 48

  // ---- input unpack (setup_inputs insertion order, nested depth-first) ----
  const int* en_input = (const int*)d_in[0];
  const int* zh_input = (const int*)d_in[1];
  // d_in[2] = teaching (unused by reference math)
  const float* en_embed = (const float*)d_in[3];
  const float* zh_embed = (const float*)d_in[4];
  const float* fc_w = (const float*)d_in[61];
  const float* fc_b = (const float*)d_in[62];

  // ---- workspace carve-out ----
  char* cur = (char*)d_ws;
  auto alloc = [&](size_t bytes) -> void* {
    void* p = (void*)cur;
    cur += (bytes + 255) & ~(size_t)255;
    return p;
  };
  float* xbuf  = (float*)alloc((size_t)ME * Dm * 4);
  float* qb    = (float*)alloc((size_t)ME * Dm * 4);
  float* kb    = (float*)alloc((size_t)ME * Dm * 4);
  float* vb    = (float*)alloc((size_t)ME * Dm * 4);
  float* t_raw = (float*)alloc((size_t)ME * Dm * 4);
  float* t_att = (float*)alloc((size_t)ME * Dm * 4);
  float* x2    = (float*)alloc((size_t)ME * Dm * 4);
  float* xn    = (float*)alloc((size_t)ME * Dm * 4);
  float* xn2   = (float*)alloc((size_t)ME * Dm * 4);
  float* ffo   = (float*)alloc((size_t)ME * Dm * 4);
  float* ffh   = (float*)alloc((size_t)ME * DFE * 4);
  float* scores= (float*)alloc((size_t)ZBH * 256 * 256 * 4);
  float* encout= (float*)alloc((size_t)ME * Dm * 4);
  float* accum = (float*)alloc(256);
  half_t* hA   = (half_t*)alloc((size_t)ME * DFE * 2);
  half_t* hB   = (half_t*)alloc((size_t)Vp64 * Dp * 2);
  half_t* hQ   = (half_t*)alloc((size_t)ZBH * 256 * HDp * 2);
  half_t* hK   = (half_t*)alloc((size_t)ZBH * 256 * HDp * 2);
  half_t* hVt  = (half_t*)alloc((size_t)ZBH * HDp * 256 * 2);
  half_t* sc16 = (half_t*)alloc((size_t)ZBH * 256 * 256 * 2);

  // ---- launch helpers ----
  auto gemm = [&](const half_t* A, long long sA, int lda,
                  const half_t* Bt, long long sB, int ldbt,
                  float* C, long long sC, int ldc,
                  int M, int N, int Kp, int batches,
                  const float* bias, float scale, int gelu, int causal,
                  int headH, int headColOff) {
    int tiles = ((M + 15) / 16) * ((N + 63) / 64);
    dim3 grid(ceilDiv(tiles, 4), 1, batches);
    wmma_gemm_kernel<<<grid, 128, 0, stream>>>(A, sA, lda, Bt, sB, ldbt, C, sC, ldc,
                                               M, N, Kp, bias, scale, gelu, causal,
                                               headH, headColOff);
  };
  auto cvtA = [&](const float* src, half_t* dst, long long rows, int cols, int Kp) {
    long long total = rows * (long long)Kp;
    cvt_a_kernel<<<ceilDiv(total, 256), 256, 0, stream>>>(src, dst, rows, cols, Kp);
  };
  auto cvtWT = [&](const float* w, half_t* dst, int K, int N, int Kp) {
    int Np = ((N + 63) / 64) * 64;  // pad Bt rows to the 64-wide wave tile
    long long total = (long long)Np * Kp;
    cvt_w_t_kernel<<<ceilDiv(total, 256), 256, 0, stream>>>(w, dst, K, N, Kp, Np);
  };
  auto addv = [&](const float* a, const float* b, float* c, long long n) {
    add_vec_kernel<<<ceilDiv(n, 256), 256, 0, stream>>>(a, b, c, n);
  };
  auto lnorm = [&](const float* x, const float* g, const float* b, float* y, long long rows) {
    layernorm_kernel<<<(int)rows, 128, 0, stream>>>(x, g, b, y, Dm);
  };

  // attention: Q-source (Lq per batch), KV-source (Lk per batch); wo output in t_att
  auto attention = [&](const float* xq, int Lq, const float* xkv, int Lk,
                       const float* wq, const float* wk, const float* wv, const float* wo,
                       int causal) {
    int Mq = Bsz * Lq, Mk = Bsz * Lk;
    int Lqp = 256, Lkp = 256;
    cvtA(xq, hA, Mq, Dm, Dp);
    cvtWT(wq, hB, Dm, Dm, Dp);
    gemm(hA, 0, Dp, hB, 0, Dp, qb, 0, Dm, Mq, Dm, Dp, 1, nullptr, 1.0f, 0, 0, 0, 0);
    cvtA(xkv, hA, Mk, Dm, Dp);
    cvtWT(wk, hB, Dm, Dm, Dp);
    gemm(hA, 0, Dp, hB, 0, Dp, kb, 0, Dm, Mk, Dm, Dp, 1, nullptr, 1.0f, 0, 0, 0, 0);
    cvtWT(wv, hB, Dm, Dm, Dp);
    gemm(hA, 0, Dp, hB, 0, Dp, vb, 0, Dm, Mk, Dm, Dp, 1, nullptr, 1.0f, 0, 0, 0, 0);
    {
      long long tQ = (long long)Bsz * Hn * Lqp * HDp;
      stage_heads_kernel<<<ceilDiv(tQ, 256), 256, 0, stream>>>(qb, hQ, Bsz, Hn, Lq, Lqp, Dm, HD, HDp);
      long long tK = (long long)Bsz * Hn * Lkp * HDp;
      stage_heads_kernel<<<ceilDiv(tK, 256), 256, 0, stream>>>(kb, hK, Bsz, Hn, Lk, Lkp, Dm, HD, HDp);
      long long tV = (long long)Bsz * Hn * HDp * Lkp;
      stage_heads_t_kernel<<<ceilDiv(tV, 256), 256, 0, stream>>>(vb, hVt, Bsz, Hn, Lk, Lkp, Dm, HD, HDp);
    }
    // scores = Q K^T / sqrt(100) (+ causal)
    gemm(hQ, (long long)Lqp * HDp, HDp,
         hK, (long long)Lkp * HDp, HDp,
         scores, 256LL * 256, 256,
         Lq, Lk, HDp, ZBH, nullptr, 0.1f, 0, causal, 0, 0);
    softmax_rows_kernel<<<dim3(256, ZBH), 256, 0, stream>>>(scores, Lq, Lk);
    {
      long long tS = (long long)ZBH * 256 * 256;
      cvt_scores16_kernel<<<ceilDiv(tS, 256), 256, 0, stream>>>(scores, sc16, tS, Lq, Lk);
    }
    // out_h = A @ V_h merged into t_raw per-head column blocks
    gemm(sc16, 256LL * 256, 256,
         hVt, (long long)HDp * Lkp, Lkp,
         t_raw, (long long)Lq * Dm, Dm,
         Lq, HD, Lkp, ZBH, nullptr, 1.0f, 0, 0, Hn, HD);
    cvtA(t_raw, hA, Mq, Dm, Dp);
    cvtWT(wo, hB, Dm, Dm, Dp);
    gemm(hA, 0, Dp, hB, 0, Dp, t_att, 0, Dm, Mq, Dm, Dp, 1, nullptr, 1.0f, 0, 0, 0, 0);
  };

  // ====================== ENCODER ======================
  {
    long long total = (long long)Bsz * Le * Dm;
    embed_pe_kernel<<<ceilDiv(total, 256), 256, 0, stream>>>(en_embed, en_input, xbuf, Bsz, 256, Le, Dm);
  }
  for (int l = 0; l < 2; ++l) {
    int e0 = 5 + 12 * l;
    const float* wq  = (const float*)d_in[e0 + 0];
    const float* wk  = (const float*)d_in[e0 + 1];
    const float* wv  = (const float*)d_in[e0 + 2];
    const float* wo  = (const float*)d_in[e0 + 3];
    const float* n1g = (const float*)d_in[e0 + 4];
    const float* n1b = (const float*)d_in[e0 + 5];
    const float* w1  = (const float*)d_in[e0 + 6];
    const float* b1  = (const float*)d_in[e0 + 7];
    const float* w2  = (const float*)d_in[e0 + 8];
    const float* b2  = (const float*)d_in[e0 + 9];
    const float* ng  = (const float*)d_in[e0 + 10];
    const float* nb  = (const float*)d_in[e0 + 11];

    attention(xbuf, Le, xbuf, Le, wq, wk, wv, wo, 0);   // -> t_att
    addv(xbuf, t_att, x2, (long long)ME * Dm);          // x = x_q + attned
    lnorm(x2, n1g, n1b, xn2, ME);                       // x_norm1
    cvtA(x2, hA, ME, Dm, Dp);
    cvtWT(w1, hB, Dm, DFE, Dp);
    gemm(hA, 0, Dp, hB, 0, Dp, ffh, 0, DFE, ME, DFE, Dp, 1, b1, 1.0f, 1, 0, 0, 0);
    cvtA(ffh, hA, ME, DFE, DFE);
    cvtWT(w2, hB, DFE, Dm, DFE);
    gemm(hA, 0, DFE, hB, 0, DFE, ffo, 0, Dm, ME, Dm, DFE, 1, b2, 1.0f, 0, 0, 0, 0);
    addv(ffo, xn2, ffo, (long long)ME * Dm);            // block_out = y + x_norm1
    addv(ffo, xbuf, x2, (long long)ME * Dm);            // + enc residual
    lnorm(x2, ng, nb, (l == 1) ? encout : xbuf, ME);    // enc = ln(...)
  }

  // ====================== DECODER ======================
  {
    long long total = (long long)Bsz * Lz * Dm;
    embed_pe_kernel<<<ceilDiv(total, 256), 256, 0, stream>>>(zh_embed, zh_input, xbuf, Bsz, 256, Lz, Dm);
  }
  for (int l = 0; l < 2; ++l) {
    int dd = 29 + 16 * l;
    const float* sawq = (const float*)d_in[dd + 0];
    const float* sawk = (const float*)d_in[dd + 1];
    const float* sawv = (const float*)d_in[dd + 2];
    const float* sawo = (const float*)d_in[dd + 3];
    const float* ng   = (const float*)d_in[dd + 4];
    const float* nb   = (const float*)d_in[dd + 5];
    const float* bwq  = (const float*)d_in[dd + 6];
    const float* bwk  = (const float*)d_in[dd + 7];
    const float* bwv  = (const float*)d_in[dd + 8];
    const float* bwo  = (const float*)d_in[dd + 9];
    const float* n1g  = (const float*)d_in[dd + 10];
    const float* n1b  = (const float*)d_in[dd + 11];
    const float* w1   = (const float*)d_in[dd + 12];
    const float* b1   = (const float*)d_in[dd + 13];
    const float* w2   = (const float*)d_in[dd + 14];
    const float* b2   = (const float*)d_in[dd + 15];

    attention(xbuf, Lz, xbuf, Lz, sawq, sawk, sawv, sawo, 1);  // masked self-attn
    addv(xbuf, t_att, x2, (long long)MZ * Dm);
    lnorm(x2, ng, nb, xn, MZ);                                 // dec2
    attention(xn, Lz, encout, Le, bwq, bwk, bwv, bwo, 0);      // cross-attn
    addv(xn, t_att, x2, (long long)MZ * Dm);                   // x = x_q + attned
    lnorm(x2, n1g, n1b, xn2, MZ);                              // x_norm1
    cvtA(x2, hA, MZ, Dm, Dp);
    cvtWT(w1, hB, Dm, DFD, Dp);
    gemm(hA, 0, Dp, hB, 0, Dp, ffh, 0, DFD, MZ, DFD, Dp, 1, b1, 1.0f, 1, 0, 0, 0);
    cvtA(ffh, hA, MZ, DFD, DFD);
    cvtWT(w2, hB, DFD, Dm, DFD);
    gemm(hA, 0, DFD, hB, 0, DFD, ffo, 0, Dm, MZ, Dm, DFD, 1, b2, 1.0f, 0, 0, 0, 0);
    addv(ffo, xn2, xbuf, (long long)MZ * Dm);                  // dec = y + x_norm1
  }

  // ====================== LOGITS / LOG-SOFTMAX / LOSS ======================
  float* out = (float*)d_out;
  cvtA(xbuf, hA, MZ, Dm, Dp);
  cvtWT(fc_w, hB, Dm, V, Dp);
  gemm(hA, 0, Dp, hB, 0, Dp, out, 0, V, MZ, V, Dp, 1, fc_b, 1.0f, 0, 0, 0, 0);

  logsoftmax_kernel<<<MZ, 256, 0, stream>>>(out, V);

  zero_acc_kernel<<<1, 32, 0, stream>>>(accum);
  nll_kernel<<<ceilDiv((long long)MZ, 256), 256, 0, stream>>>(out, zh_input, accum, Bsz, Lz, 256, V);
  finish_loss_kernel<<<1, 32, 0, stream>>>(accum, out + (long long)MZ * V);
}
